// MNISTBaseline_17257178595518
// MI455X (gfx1250) — matrix operand order repaired and checked by reading, not compiled, not run on hardware
//
#include <hip/hip_runtime.h>

// ---------------- problem constants ----------------
#define T_STEPS 784
#define BATCH   128
#define HID     256
#define EMB     128
#define IN0     384
#define FOURH   1024
#define LUTROWS 58            // 28 x-rows + 28 y-rows + 2 p-rows

// ---------------- partitioning ----------------
#define NWG     16            // persistent workgroups
#define CPW     16            // h-columns per WG  (HID / NWG)
#define GROWS   64            // gate rows per WG  (4 * CPW)
#define THREADS 256           // 8 waves of 32
#define WAVES   8

#define AS1 __attribute__((address_space(1)))
#define AS3 __attribute__((address_space(3)))

// ---- probe-guarded gfx1250 async / monitor paths ----
#ifdef __has_builtin
#  if __has_builtin(__builtin_amdgcn_global_load_async_to_lds_b128)
#    define USE_ASYNC_LDS 1
#  endif
#  if __has_builtin(__builtin_amdgcn_s_wait_asynccnt)
#    define HAVE_WAIT_ASYNC 1
#  endif
#  if __has_builtin(__builtin_amdgcn_global_load_monitor_b32) && \
      __has_builtin(__builtin_amdgcn_s_monitor_sleep)
#    define USE_MONITOR 1
#  endif
#endif

typedef unsigned short u16;
typedef __attribute__((ext_vector_type(16))) __bf16 v16bf;
typedef __attribute__((ext_vector_type(8)))  float  v8f;
typedef __attribute__((ext_vector_type(4)))  int    v4i;

union V8U  { v8f  v; float s[8]; };
union V16U { v16bf v; uint4 q[2]; };

__device__ __forceinline__ u16 f2bf(float f) {
  union { float f; unsigned u; } v; v.f = f;
  unsigned r = v.u + 0x7FFFu + ((v.u >> 16) & 1u);   // round-to-nearest-even
  return (u16)(r >> 16);
}
__device__ __forceinline__ float bf2f(u16 x) {
  union { unsigned u; float f; } v; v.u = ((unsigned)x) << 16; return v.f;
}
__device__ __forceinline__ float sigm(float x) { return 1.0f / (1.0f + __expf(-x)); }

// A operand: 16x32 bf16 tile from row-major [*, HID] bf16 matrix.
// ISA layout: lane L holds M = L&15; lanes 0-15 K={0..7,16..23}, lanes 16-31 K={8..15,24..31}.
__device__ __forceinline__ v16bf loadA(const u16* __restrict__ base, int row0, int k0, int lane) {
  const u16* p = base + (size_t)(row0 + (lane & 15)) * HID + k0 + ((lane >> 4) << 3);
  V16U u;
  u.q[0] = *(const uint4*)(p);
  u.q[1] = *(const uint4*)(p + 16);
  return u.v;
}

// B operand: 32x16 bf16 tile.  Weight slice stored [16 rows][HID] row-major (K contiguous),
// so lane L (N = L&15) reads K = k0 + 8*(L>>4) + {0..7, 16..23}.
__device__ __forceinline__ v16bf loadB(const u16* __restrict__ wrow, int k0, int lane) {
  const u16* p = wrow + (size_t)(lane & 15) * HID + k0 + ((lane >> 4) << 3);
  V16U u;
  u.q[0] = *(const uint4*)(p);
  u.q[1] = *(const uint4*)(p + 16);
  return u.v;
}

__device__ __forceinline__ v8f wmma_bf16(v16bf a, v16bf b, v8f c) {
  return __builtin_amdgcn_wmma_f32_16x16x32_bf16(false, a, false, b, (short)0, c, false, false);
}

// 16-byte global -> LDS copy; async path on gfx1250, VGPR round-trip otherwise.
__device__ __forceinline__ void copy16_g2l(const uint4* gsrc, uint4* ldst) {
#if USE_ASYNC_LDS
  __builtin_amdgcn_global_load_async_to_lds_b128((AS1 v4i*)gsrc, (AS3 v4i*)ldst, 0, 0);
#else
  *ldst = *gsrc;
#endif
}
__device__ __forceinline__ void wait_async_copies() {
#if USE_ASYNC_LDS
#  if HAVE_WAIT_ASYNC
  __builtin_amdgcn_s_wait_asynccnt(0);
#  else
  asm volatile("s_wait_asynccnt 0" ::: "memory");
#  endif
#endif
}

// ---------------- prep kernel: bf16 weights, LUTs, init state, reset sync ----------------
#define N_CONV (5 * FOURH * HID)
#define N_LUT  (LUTROWS * FOURH)
#define N_BIAS (2 * FOURH)
#define N_HB   (3 * BATCH * HID)
#define N_TOT  (N_CONV + N_LUT + N_BIAS + N_HB)

__global__ void lstm_prep_kernel(
    const float* __restrict__ Whh0, const float* __restrict__ Whh1, const float* __restrict__ Whh2,
    const float* __restrict__ Wih1, const float* __restrict__ Wih2,
    const float* __restrict__ Wih0,
    const float* __restrict__ x_emb, const float* __restrict__ y_emb, const float* __restrict__ p_emb,
    const float* __restrict__ bih0, const float* __restrict__ bhh0,
    const float* __restrict__ bih1, const float* __restrict__ bhh1,
    const float* __restrict__ bih2, const float* __restrict__ bhh2,
    const float* __restrict__ h0,
    u16* __restrict__ wAll, float* __restrict__ lut, float* __restrict__ bias12,
    u16* __restrict__ hbuf, unsigned* __restrict__ syncc)
{
  int id = blockIdx.x * THREADS + threadIdx.x;
  if (id == 0) *syncc = 0u;                       // reset grid-sync counter every call
  if (id < N_CONV) {
    int m = id / (FOURH * HID);
    int rem = id - m * (FOURH * HID);
    const float* src = (m == 0) ? Whh0 : (m == 1) ? Whh1 : (m == 2) ? Whh2
                     : (m == 3) ? Wih1 : Wih2;
    wAll[id] = f2bf(src[rem]);
    return;
  }
  id -= N_CONV;
  if (id < N_LUT) {
    int c   = id & (FOURH - 1);
    int row = id >> 10;
    const float* emb; int off;
    if (row < 28)      { emb = x_emb + (size_t)row * EMB;        off = 0; }
    else if (row < 56) { emb = y_emb + (size_t)(row - 28) * EMB; off = EMB; }
    else               { emb = p_emb + (size_t)(row - 56) * EMB; off = 2 * EMB; }
    const float* w = Wih0 + (size_t)c * IN0 + off;
    float s = 0.f;
    #pragma unroll 8
    for (int k = 0; k < EMB; ++k) s += emb[k] * w[k];
    if (row >= 56) s += bih0[c] + bhh0[c];        // fold layer-0 bias into p-LUT
    lut[(size_t)row * FOURH + c] = s;
    return;
  }
  id -= N_LUT;
  if (id < N_BIAS) {
    int l = id >> 10, c = id & (FOURH - 1);
    bias12[id] = (l ? bih2[c] + bhh2[c] : bih1[c] + bhh1[c]);
    return;
  }
  id -= N_BIAS;
  if (id < N_HB) {
    int l = id / (BATCH * HID);
    int rem = id - l * (BATCH * HID);
    int b = rem >> 8, n = rem & (HID - 1);
    // buffer 0 of each layer holds h_{t=-1}
    hbuf[(size_t)(l * 2 + 0) * BATCH * HID + (size_t)b * HID + n] = f2bf(h0[l * HID + n]);
  }
}

// ---------------- persistent recurrent kernel ----------------
__device__ __forceinline__ void grid_sync(unsigned* cnt, unsigned target) {
  __threadfence();
  __syncthreads();
  if (threadIdx.x == 0) {
    atomicAdd(cnt, 1u);
#if USE_MONITOR
    // L2 memory-monitor wait: wave sleeps until the counter cacheline is written.
    while ((unsigned)__builtin_amdgcn_global_load_monitor_b32((AS1 int*)cnt, 0, 0) < target)
      __builtin_amdgcn_s_monitor_sleep(32);
#else
    while (*(volatile unsigned*)cnt < target) __builtin_amdgcn_s_sleep(2);
#endif
  }
  __syncthreads();
  __threadfence();
}

__global__ void __launch_bounds__(THREADS, 1)
lstm_persistent_kernel(
    const int*   __restrict__ inputs,   // [T,B,3]
    const float* __restrict__ c0,       // [3,H]
    const float* __restrict__ Wout,     // [1,H]
    const float* __restrict__ bout,     // [1]
    const u16*   __restrict__ wAll,     // bf16 {Whh0,Whh1,Whh2,Wih1,Wih2} each [4H,H]
    const float* __restrict__ lut,      // [58,4H]
    const float* __restrict__ bias12,   // [2,4H]
    u16*         __restrict__ hbuf,     // [3][2][B][H] bf16
    unsigned*    __restrict__ syncc,
    float*       __restrict__ out)      // outs[T,B] ++ hT[3,B,H] ++ cT[3,B,H]
{
  extern __shared__ char smem[];
  u16*   sWhh  = (u16*)smem;                            // [3][GROWS][HID]
  u16*   sWih  = sWhh + 3 * GROWS * HID;                // [2][GROWS][HID]
  float* sLut  = (float*)(sWih + 2 * GROWS * HID);      // [58][GROWS]
  float* sBias = sLut + LUTROWS * GROWS;                // [2][GROWS]

  const int tid  = threadIdx.x;
  const int lane = tid & 31;
  const int wave = tid >> 5;
  const int wg   = blockIdx.x;
  const int colbase = wg * CPW;

  // ---- stage this WG's weight column slice into LDS via async copies ----
  // (lives in LDS for all 784 steps; max 40 outstanding async ops/thread < 63)
  for (int m = 0; m < 5; ++m) {
    const u16* src = wAll + (size_t)m * FOURH * HID;
    u16* dst = (m < 3) ? (sWhh + m * GROWS * HID) : (sWih + (m - 3) * GROWS * HID);
    for (int c = tid; c < GROWS * (HID / 8); c += THREADS) {     // 8 bf16 per uint4
      int r = c >> 5, q = c & 31;
      int grow = (r >> 4) * HID + colbase + (r & 15);            // gate*H + col
      copy16_g2l(((const uint4*)(src + (size_t)grow * HID)) + q,
                 ((uint4*)(dst + (size_t)r * HID)) + q);
    }
  }
  for (int e = tid; e < LUTROWS * GROWS; e += THREADS) {
    int row = e / GROWS, j = e & (GROWS - 1);
    int grow = (j >> 4) * HID + colbase + (j & 15);
    sLut[e] = lut[(size_t)row * FOURH + grow];
  }
  for (int e = tid; e < 2 * GROWS; e += THREADS) {
    int l = e >> 6, j = e & (GROWS - 1);
    int grow = (j >> 4) * HID + colbase + (j & 15);
    sBias[e] = bias12[(size_t)l * FOURH + grow];
  }
  wait_async_copies();
  __syncthreads();

  // ---- cell state lives in VGPRs for the entire sequence ----
  const int n   = lane & 15;                      // local column
  const int bq  = wave * 16 + ((lane >> 4) << 3); // batch base of this lane's C rows
  float cacc[3][8];
  #pragma unroll
  for (int l = 0; l < 3; ++l)
    #pragma unroll
    for (int r = 0; r < 8; ++r)
      cacc[l][r] = c0[l * HID + colbase + n];

  unsigned syncN = 0;

  for (int t = 0; t < T_STEPS; ++t) {
    const int rd = t & 1, wr = rd ^ 1;
    if (t + 1 < T_STEPS)
      __builtin_prefetch(inputs + (size_t)(t + 1) * BATCH * 3, 0, 1);  // global_prefetch_b8

    for (int l = 0; l < 3; ++l) {
      v8f g0 = {}, g1 = {}, g2 = {}, g3 = {};

      // recurrent GEMM: h_{l,t-1} @ Whh_l^T   (K = 256, bf16 WMMA, B from LDS)
      {
        const u16* sW = sWhh + (size_t)l * GROWS * HID;
        const u16* hA = hbuf + (size_t)(l * 2 + rd) * BATCH * HID;
        #pragma unroll
        for (int kt = 0; kt < 8; ++kt) {
          v16bf a = loadA(hA, wave * 16, kt * 32, lane);
          g0 = wmma_bf16(a, loadB(sW + 0 * 16 * HID, kt * 32, lane), g0);
          g1 = wmma_bf16(a, loadB(sW + 1 * 16 * HID, kt * 32, lane), g1);
          g2 = wmma_bf16(a, loadB(sW + 2 * 16 * HID, kt * 32, lane), g2);
          g3 = wmma_bf16(a, loadB(sW + 3 * 16 * HID, kt * 32, lane), g3);
        }
      }
      // input GEMM for layers 1,2: h_{l-1,t} @ Wih_l^T
      if (l > 0) {
        const u16* sX = sWih + (size_t)(l - 1) * GROWS * HID;
        const u16* xA = hbuf + (size_t)((l - 1) * 2 + wr) * BATCH * HID;
        #pragma unroll
        for (int kt = 0; kt < 8; ++kt) {
          v16bf a = loadA(xA, wave * 16, kt * 32, lane);
          g0 = wmma_bf16(a, loadB(sX + 0 * 16 * HID, kt * 32, lane), g0);
          g1 = wmma_bf16(a, loadB(sX + 1 * 16 * HID, kt * 32, lane), g1);
          g2 = wmma_bf16(a, loadB(sX + 2 * 16 * HID, kt * 32, lane), g2);
          g3 = wmma_bf16(a, loadB(sX + 3 * 16 * HID, kt * 32, lane), g3);
        }
      }

      V8U ui, uf, ug, uo;
      ui.v = g0; uf.v = g1; ug.v = g2; uo.v = g3;

      if (l == 0) {
        // layer-0 input contribution folded into 3 LUT gathers (bias already in p-LUT)
        const int* it = inputs + (size_t)t * BATCH * 3;
        #pragma unroll
        for (int r = 0; r < 8; ++r) {
          int b = bq + r;
          int ix = it[3 * b + 0], iy = it[3 * b + 1], ip = it[3 * b + 2];
          const float* lx = sLut + ix * GROWS;
          const float* ly = sLut + (28 + iy) * GROWS;
          const float* lp = sLut + (56 + ip) * GROWS;
          ui.s[r] += lx[0 * 16 + n] + ly[0 * 16 + n] + lp[0 * 16 + n];
          uf.s[r] += lx[1 * 16 + n] + ly[1 * 16 + n] + lp[1 * 16 + n];
          ug.s[r] += lx[2 * 16 + n] + ly[2 * 16 + n] + lp[2 * 16 + n];
          uo.s[r] += lx[3 * 16 + n] + ly[3 * 16 + n] + lp[3 * 16 + n];
        }
      } else {
        const float* bs = sBias + (l - 1) * GROWS;
        #pragma unroll
        for (int r = 0; r < 8; ++r) {
          ui.s[r] += bs[0 * 16 + n];
          uf.s[r] += bs[1 * 16 + n];
          ug.s[r] += bs[2 * 16 + n];
          uo.s[r] += bs[3 * 16 + n];
        }
      }

      // LSTM cell + write this WG's h-column slice (bf16) for broadcast via L2
      u16* hw = hbuf + (size_t)(l * 2 + wr) * BATCH * HID;
      #pragma unroll
      for (int r = 0; r < 8; ++r) {
        float iv = sigm(ui.s[r]);
        float fv = sigm(uf.s[r]);
        float gv = tanhf(ug.s[r]);
        float ov = sigm(uo.s[r]);
        float cv = fv * cacc[l][r] + iv * gv;
        cacc[l][r] = cv;
        float hv = ov * tanhf(cv);
        hw[(size_t)(bq + r) * HID + colbase + n] = f2bf(hv);
      }

      grid_sync(syncc, NWG * (++syncN));
    }

    // projection: out[t,b] = h2 . Wout + bout (WG0 only; h2 is L2-resident, 64 KB)
    if (wg == 0 && tid < BATCH) {
      const u16* h2 = hbuf + (size_t)(2 * 2 + wr) * BATCH * HID + (size_t)tid * HID;
      float s = bout[0];
      #pragma unroll 8
      for (int k = 0; k < HID; ++k) s += bf2f(h2[k]) * Wout[k];
      out[(size_t)t * BATCH + tid] = s;
    }
  }

  // ---- final states ----
  const int fin = T_STEPS & 1;                 // buffer holding h_{T-1}
  float* hT = out + (size_t)T_STEPS * BATCH;
  float* cT = hT + (size_t)3 * BATCH * HID;
  for (int e = tid; e < 3 * BATCH * CPW; e += THREADS) {
    int l = e / (BATCH * CPW);
    int rem = e - l * (BATCH * CPW);
    int b = rem >> 4, j = rem & 15;
    int col = colbase + j;
    hT[((size_t)l * BATCH + b) * HID + col] =
        bf2f(hbuf[(size_t)(l * 2 + fin) * BATCH * HID + (size_t)b * HID + col]);
  }
  #pragma unroll
  for (int l = 0; l < 3; ++l)
    #pragma unroll
    for (int r = 0; r < 8; ++r)
      cT[((size_t)l * BATCH + (bq + r)) * HID + colbase + n] = cacc[l][r];
}

// ---------------- host launch ----------------
extern "C" void kernel_launch(void* const* d_in, const int* in_sizes, int n_in,
                              void* d_out, int out_size, void* d_ws, size_t ws_size,
                              hipStream_t stream) {
  const int*   inputs = (const int*)  d_in[0];
  const float* x_emb  = (const float*)d_in[1];
  const float* y_emb  = (const float*)d_in[2];
  const float* p_emb  = (const float*)d_in[3];
  const float* Wih0   = (const float*)d_in[4];
  const float* Whh0   = (const float*)d_in[5];
  const float* bih0   = (const float*)d_in[6];
  const float* bhh0   = (const float*)d_in[7];
  const float* Wih1   = (const float*)d_in[8];
  const float* Whh1   = (const float*)d_in[9];
  const float* bih1   = (const float*)d_in[10];
  const float* bhh1   = (const float*)d_in[11];
  const float* Wih2   = (const float*)d_in[12];
  const float* Whh2   = (const float*)d_in[13];
  const float* bih2   = (const float*)d_in[14];
  const float* bhh2   = (const float*)d_in[15];
  const float* h0     = (const float*)d_in[16];
  const float* c0     = (const float*)d_in[17];
  const float* Wout   = (const float*)d_in[18];
  const float* bout   = (const float*)d_in[19];

  char* ws = (char*)d_ws;
  size_t o = 0;
  u16*      wAll   = (u16*)(ws + o);   o += (size_t)5 * FOURH * HID * sizeof(u16);
  float*    lut    = (float*)(ws + o); o += (size_t)LUTROWS * FOURH * sizeof(float);
  float*    bias12 = (float*)(ws + o); o += (size_t)2 * FOURH * sizeof(float);
  u16*      hbuf   = (u16*)(ws + o);   o += (size_t)6 * BATCH * HID * sizeof(u16);
  unsigned* syncc  = (unsigned*)(ws + o);

  int prep_blocks = (N_TOT + THREADS - 1) / THREADS;
  lstm_prep_kernel<<<prep_blocks, THREADS, 0, stream>>>(
      Whh0, Whh1, Whh2, Wih1, Wih2, Wih0,
      x_emb, y_emb, p_emb,
      bih0, bhh0, bih1, bhh1, bih2, bhh2,
      h0, wAll, lut, bias12, hbuf, syncc);

  size_t smem = (size_t)(3 * GROWS * HID + 2 * GROWS * HID) * sizeof(u16)
              + (size_t)LUTROWS * GROWS * sizeof(float)
              + (size_t)2 * GROWS * sizeof(float);   // 179,200 B < 320 KB WGP LDS

  lstm_persistent_kernel<<<dim3(NWG), dim3(THREADS), smem, stream>>>(
      inputs, c0, Wout, bout, wAll, lut, bias12, hbuf, syncc, (float*)d_out);
}
